// PointNetPPSeg_17841294147733
// MI455X (gfx1250) — matrix operand order repaired and checked by reading, not compiled, not run on hardware
//
#include <hip/hip_runtime.h>
#include <hip/hip_bf16.h>

typedef __attribute__((ext_vector_type(16))) _Float16 v16h;
typedef __attribute__((ext_vector_type(8)))  _Float16 v8h;
typedef __attribute__((ext_vector_type(8)))  float    v8f;

#define BATCH 8
#define NPTS  8192
#define NSAMP 32
#define NCLS  50

// ---------------------------------------------------------------------------
// Farthest point sampling: one workgroup per batch, dists resident in LDS.
// ---------------------------------------------------------------------------
__global__ void fps_kernel(const float* __restrict__ xyz, int* __restrict__ out,
                           int n, int npoint) {
  __shared__ float s_d[8192];
  __shared__ float s_rv[256];
  __shared__ int   s_ri[256];
  __shared__ int   s_cur;
  const int b = blockIdx.x, t = threadIdx.x;
  const float* p = xyz + (size_t)b * n * 3;
  for (int j = t; j < n; j += 256) s_d[j] = 1e10f;
  if (t == 0) s_cur = 0;
  __syncthreads();
  for (int it = 0; it < npoint; ++it) {
    const int cur = s_cur;
    if (t == 0) out[b * npoint + it] = cur;
    const float cx = p[cur * 3 + 0], cy = p[cur * 3 + 1], cz = p[cur * 3 + 2];
    float bv = -1.0f; int bi = 0;
    for (int j = t; j < n; j += 256) {
      const float dx = p[j * 3 + 0] - cx;
      const float dy = p[j * 3 + 1] - cy;
      const float dz = p[j * 3 + 2] - cz;
      const float d = dx * dx + dy * dy + dz * dz;
      const float m = fminf(s_d[j], d);
      s_d[j] = m;
      if (m > bv) { bv = m; bi = j; }
    }
    s_rv[t] = bv; s_ri[t] = bi;
    __syncthreads();
    for (int off = 128; off > 0; off >>= 1) {
      if (t < off && s_rv[t + off] > s_rv[t]) {
        s_rv[t] = s_rv[t + off]; s_ri[t] = s_ri[t + off];
      }
      __syncthreads();
    }
    if (t == 0) s_cur = s_ri[0];
    __syncthreads();
  }
}

__global__ void gather_xyz_kernel(const float* __restrict__ xyz, const int* __restrict__ idx,
                                  float* __restrict__ out, int n, int P) {
  const int tid = blockIdx.x * blockDim.x + threadIdx.x;
  if (tid >= BATCH * P) return;
  const int b = tid / P;
  const int i = idx[tid];
  const float* s = xyz + ((size_t)b * n + i) * 3;
  out[tid * 3 + 0] = s[0];
  out[tid * 3 + 1] = s[1];
  out[tid * 3 + 2] = s[2];
}

__global__ void ball_query_kernel(const float* __restrict__ new_xyz,
                                  const float* __restrict__ xyz,
                                  int* __restrict__ out, int n, int P, float r2) {
  const int tid = blockIdx.x * blockDim.x + threadIdx.x;
  if (tid >= BATCH * P) return;
  const int b = tid / P;
  const float* q = new_xyz + (size_t)tid * 3;
  const float* src = xyz + (size_t)b * n * 3;
  int* o = out + (size_t)tid * NSAMP;
  int cnt = 0;
  for (int j = 0; j < n && cnt < NSAMP; ++j) {
    const float dx = src[j * 3 + 0] - q[0];
    const float dy = src[j * 3 + 1] - q[1];
    const float dz = src[j * 3 + 2] - q[2];
    if (dx * dx + dy * dy + dz * dz < r2) o[cnt++] = j;
  }
  const int first = (cnt > 0) ? o[0] : 0;
  for (int k = cnt; k < NSAMP; ++k) o[k] = first;
}

// ---------------------------------------------------------------------------
// Weight packing: f32 [Cout x Cin] -> zero-padded f16 [Coutp x Cinp] (+bias pad)
// ---------------------------------------------------------------------------
__global__ void pack_w_kernel(const float* __restrict__ W, const float* __restrict__ b,
                              _Float16* __restrict__ Wp, float* __restrict__ bp,
                              int Cin, int Cout, int Cinp, int Coutp) {
  const int tid = blockIdx.x * blockDim.x + threadIdx.x;
  const int total = Coutp * Cinp;
  if (tid < total) {
    const int r = tid / Cinp, c = tid % Cinp;
    Wp[tid] = (r < Cout && c < Cin) ? (_Float16)W[(size_t)r * Cin + c] : (_Float16)0.0f;
  }
  if (tid < Coutp) bp[tid] = (b != nullptr && tid < Cout) ? b[tid] : 0.0f;
}

// ---------------------------------------------------------------------------
// Grouped input builder. X is point-major f16 [Mtot x Cinp]:
// per sample: [dx,dy,dz, feats(Cf), zeros(pad)].
// feats is point-major f16 [B*n x Cf] (Cf already a multiple of 32).
// ---------------------------------------------------------------------------
__global__ void group_kernel(const float* __restrict__ xyz, const _Float16* __restrict__ feats,
                             const float* __restrict__ nxyz, const int* __restrict__ idx,
                             _Float16* __restrict__ X, int n, int P, int Cf, int Cinp) {
  const int Mtot = BATCH * P * NSAMP;
  const int m = blockIdx.x * blockDim.x + threadIdx.x;
  if (m >= Mtot) return;
  const int bp = m / NSAMP;
  const int b = bp / P;
  const int i = idx[m];
  const float* s = xyz + ((size_t)b * n + i) * 3;
  const float* q = nxyz + (size_t)bp * 3;
  _Float16* xc = X + (size_t)m * Cinp;
  xc[0] = (_Float16)(s[0] - q[0]);
  xc[1] = (_Float16)(s[1] - q[1]);
  xc[2] = (_Float16)(s[2] - q[2]);
  const _Float16* fr = feats + ((size_t)b * n + i) * Cf;  // ok if Cf==0
  for (int c = 0; c < Cf; ++c) xc[3 + c] = fr[c];
  for (int c = 3 + Cf; c < Cinp; ++c) xc[c] = (_Float16)0.0f;
}

// Max over the NSAMP window; Y point-major f16 [BP*NSAMP x C] -> out [BP x C].
__global__ void maxpool_kernel(const _Float16* __restrict__ Y, _Float16* __restrict__ out,
                               int C, int BP) {
  const int tid = blockIdx.x * blockDim.x + threadIdx.x;
  if (tid >= BP * C) return;
  const int bp = tid / C, c = tid % C;
  const _Float16* p = Y + (size_t)bp * NSAMP * C + c;
  float m = (float)p[0];
  for (int s = 1; s < NSAMP; ++s) m = fmaxf(m, (float)p[(size_t)s * C]);
  out[(size_t)bp * C + c] = (_Float16)m;
}

__global__ void three_nn_kernel(const float* __restrict__ unk, const float* __restrict__ kn,
                                int* __restrict__ oidx, float* __restrict__ ow,
                                int nu, int nk) {
  const int tid = blockIdx.x * blockDim.x + threadIdx.x;
  if (tid >= BATCH * nu) return;
  const int b = tid / nu;
  const float* q = unk + (size_t)tid * 3;
  const float* src = kn + (size_t)b * nk * 3;
  float d0 = 1e30f, d1 = 1e30f, d2 = 1e30f;
  int i0 = 0, i1 = 0, i2 = 0;
  for (int j = 0; j < nk; ++j) {
    const float dx = src[j * 3 + 0] - q[0];
    const float dy = src[j * 3 + 1] - q[1];
    const float dz = src[j * 3 + 2] - q[2];
    const float d = dx * dx + dy * dy + dz * dz;
    if (d < d0)      { d2 = d1; i2 = i1; d1 = d0; i1 = i0; d0 = d; i0 = j; }
    else if (d < d1) { d2 = d1; i2 = i1; d1 = d;  i1 = j; }
    else if (d < d2) { d2 = d;  i2 = j; }
  }
  const float r0 = 1.0f / (d0 + 1e-8f);
  const float r1 = 1.0f / (d1 + 1e-8f);
  const float r2 = 1.0f / (d2 + 1e-8f);
  const float s = r0 + r1 + r2;
  oidx[tid * 3 + 0] = i0; oidx[tid * 3 + 1] = i1; oidx[tid * 3 + 2] = i2;
  ow[tid * 3 + 0] = r0 / s; ow[tid * 3 + 1] = r1 / s; ow[tid * 3 + 2] = r2 / s;
}

// Interpolate known feats then append skip feats; all point-major f16.
// X row = [interp(Ck), skip(Cu)]; Ck+Cu is a multiple of 32.
__global__ void interp_concat_kernel(const _Float16* __restrict__ kf, const _Float16* __restrict__ uf,
                                     const int* __restrict__ idx, const float* __restrict__ w,
                                     _Float16* __restrict__ X, int nu, int nk, int Ck, int Cu) {
  const int M = BATCH * nu;
  const int col = blockIdx.x * blockDim.x + threadIdx.x;
  if (col >= M) return;
  const int b = col / nu;
  const size_t base = (size_t)b * nk;
  const _Float16* r0 = kf + (base + idx[col * 3 + 0]) * Ck;
  const _Float16* r1 = kf + (base + idx[col * 3 + 1]) * Ck;
  const _Float16* r2 = kf + (base + idx[col * 3 + 2]) * Ck;
  const float w0 = w[col * 3 + 0], w1 = w[col * 3 + 1], w2 = w[col * 3 + 2];
  _Float16* xc = X + (size_t)col * (Ck + Cu);
  for (int c = 0; c < Ck; ++c)
    xc[c] = (_Float16)(w0 * (float)r0[c] + w1 * (float)r1[c] + w2 * (float)r2[c]);
  const _Float16* ur = uf + (size_t)col * Cu;  // ok if Cu==0
  for (int c = 0; c < Cu; ++c) xc[Ck + c] = ur[c];
}

// ---------------------------------------------------------------------------
// WMMA pointwise-MLP GEMM: Y[m][co] = act( sum_k Wp[co][k] * X[m][k] + bp[co] )
// Wp: f16 [Coutp x Cinp] zero-padded; X: f16 point-major [M x Cinp];
// Y: f16 point-major [M x Coutp]. One wave = one 16x16 tile; 4 waves/block.
// Inner loop: 4x global_load_b128 + 1x v_wmma_f32_16x16x32_f16, no predicates.
// ---------------------------------------------------------------------------
__global__ void wmma_mlp_kernel(const _Float16* __restrict__ Wp, const float* __restrict__ bp,
                                const _Float16* __restrict__ X, _Float16* __restrict__ Y,
                                int Cinp, int Coutp, int relu) {
  const int lane = threadIdx.x & 31;
  const int wave = threadIdx.x >> 5;
  const int colB = (blockIdx.x * 4 + wave) * 16 + (lane & 15);  // point index
  const int co0 = blockIdx.y * 16;                              // channel tile
  const bool lo = (lane < 16);
  const int rowA = co0 + (lane & 15);
  const _Float16* wrow = Wp + (size_t)rowA * Cinp;
  const _Float16* xcol = X + (size_t)colB * Cinp;
  // Documented 16-bit fragment layouts (CDNA5 ISA 7.12.2):
  // A 16x32: lanes<16 hold K {0..7,16..23}; lanes>=16 hold K {8..15,24..31}
  // B 32x16: lanes<16 hold K {0..7,16..23}; lanes>=16 hold K {8..15,24..31}
  const int aoff0 = lo ? 0 : 8;
  const int aoff1 = lo ? 16 : 24;
  const int boff  = lo ? 0 : 16;
  v8f acc = {};
  for (int k0 = 0; k0 < Cinp; k0 += 32) {
    const v8h a0 = *(const v8h*)(wrow + k0 + aoff0);
    const v8h a1 = *(const v8h*)(wrow + k0 + aoff1);
    const v8h b0 = *(const v8h*)(xcol + k0 + aoff0);
    const v8h b1 = *(const v8h*)(xcol + k0 + aoff1);
    __builtin_prefetch(xcol + k0 + 64, 0, 0);  // global_prefetch_b8, speculative
    const v16h afrag = __builtin_shufflevector(a0, a1, 0, 1, 2, 3, 4, 5, 6, 7,
                                               8, 9, 10, 11, 12, 13, 14, 15);
    const v16h bfrag = __builtin_shufflevector(b0, b1, 0, 1, 2, 3, 4, 5, 6, 7,
                                               8, 9, 10, 11, 12, 13, 14, 15);
    acc = __builtin_amdgcn_wmma_f32_16x16x32_f16(false, afrag, false, bfrag,
                                                 (short)0, acc, false, false);
  }
  // C/D layout: VGPR r holds channel co0+r (lanes 0-15) / co0+r+8 (lanes 16-31)
  const int rbase = co0 + (lo ? 0 : 8);
  v8h outv;
#pragma unroll
  for (int r = 0; r < 8; ++r) {
    float v = acc[r] + bp[rbase + r];
    if (relu) v = fmaxf(v, 0.0f);
    outv[r] = (_Float16)v;
  }
  *(v8h*)(Y + (size_t)colB * Coutp + rbase) = outv;
}

// f16 [M x Cp] (Cp=64, first NCLS valid) -> f32 [M x NCLS]
__global__ void head_transpose_kernel(const _Float16* __restrict__ Y, float* __restrict__ out,
                                      int M, int Cp) {
  const int tid = blockIdx.x * blockDim.x + threadIdx.x;
  if (tid >= M * NCLS) return;
  const int m = tid / NCLS, o = tid % NCLS;
  out[tid] = (float)Y[(size_t)m * Cp + o];
}

// ---------------------------------------------------------------------------
static inline int ceil_div(int a, int b) { return (a + b - 1) / b; }

static void run_mlp(const _Float16* Wp, const float* bp, const _Float16* X, _Float16* Y,
                    int Cinp, int Coutp, int M, int relu, hipStream_t stream) {
  dim3 grid(M / 64, Coutp / 16);
  wmma_mlp_kernel<<<grid, dim3(128), 0, stream>>>(Wp, bp, X, Y, Cinp, Coutp, relu);
}

extern "C" void kernel_launch(void* const* d_in, const int* in_sizes, int n_in,
                              void* d_out, int out_size, void* d_ws, size_t ws_size,
                              hipStream_t stream) {
  (void)in_sizes; (void)n_in; (void)out_size; (void)ws_size;

  // ---- inputs, setup_inputs() dict (insertion) order ----
  const float* pc = (const float*)d_in[0];  // (8, 8192, 3)
  const float* saW[4][3]; const float* saB[4][3];
  int di = 1;
  for (int i = 0; i < 4; ++i)
    for (int l = 0; l < 3; ++l) {
      saW[i][l] = (const float*)d_in[di++];
      saB[i][l] = (const float*)d_in[di++];
    }
  const int fpNL[4] = {3, 2, 2, 2};
  const float* fpW[4][3]; const float* fpB[4][3];
  for (int i = 0; i < 4; ++i)
    for (int l = 0; l < fpNL[i]; ++l) {
      fpW[i][l] = (const float*)d_in[di++];
      fpB[i][l] = (const float*)d_in[di++];
    }
  const float* fc1w = (const float*)d_in[di++];
  const float* fc2w = (const float*)d_in[di++];
  const float* fc2b = (const float*)d_in[di++];

  // ---- network shapes ----
  const int NP[4] = {1024, 256, 64, 16};
  const float RAD[4] = {0.1f, 0.2f, 0.4f, 0.8f};
  const int saCin[4][3]  = {{3, 32, 32}, {67, 64, 64}, {131, 128, 128}, {259, 256, 256}};
  const int saCinP[4]    = {32, 96, 160, 288};  // layer-0 padded K (xyz+feats)
  const int saCout[4][3] = {{32, 32, 64}, {64, 64, 128}, {128, 128, 256}, {256, 256, 512}};
  const int fpSpec[4][4] = {{128, 128, 128, 128}, {320, 256, 128, 0},
                            {384, 256, 256, 0},   {768, 256, 256, 0}};

  // ---- workspace bump allocator ----
  char* wp_base = (char*)d_ws;
  size_t off = 0;
  auto alloc = [&](size_t bytes) -> void* {
    void* p = wp_base + off;
    off = (off + bytes + 255) & ~(size_t)255;
    return p;
  };
  _Float16* bufA = (_Float16*)alloc((size_t)20 << 20);
  _Float16* bufB = (_Float16*)alloc((size_t)36 << 20);
  int* fpsidx  = (int*)alloc((size_t)BATCH * 1024 * sizeof(int));
  int* ballidx = (int*)alloc((size_t)BATCH * 1024 * NSAMP * sizeof(int));
  float* xyzl[4]; _Float16* featl[4]; _Float16* fpf[4];
  for (int i = 0; i < 4; ++i)
    xyzl[i] = (float*)alloc((size_t)BATCH * NP[i] * 3 * sizeof(float));
  for (int i = 0; i < 4; ++i)
    featl[i] = (_Float16*)alloc((size_t)BATCH * NP[i] * saCout[i][2] * sizeof(_Float16));
  for (int lvl = 0; lvl < 4; ++lvl) {
    const int nu = (lvl == 0) ? NPTS : NP[lvl - 1];
    fpf[lvl] = (_Float16*)alloc((size_t)BATCH * nu * fpSpec[lvl][fpNL[lvl]] * sizeof(_Float16));
  }
  int*   nnidx = (int*)alloc((size_t)BATCH * NPTS * 3 * sizeof(int));
  float* nnw   = (float*)alloc((size_t)BATCH * NPTS * 3 * sizeof(float));

  // ---- pack all weights to padded f16 (runs first on the same stream) ----
  auto pack = [&](const float* W, const float* b, int Cin, int Cout, int Cinp, int Coutp,
                  _Float16** Wp, float** bp) {
    *Wp = (_Float16*)alloc((size_t)Coutp * Cinp * sizeof(_Float16));
    *bp = (float*)alloc((size_t)Coutp * sizeof(float));
    const int total = Coutp * Cinp;
    pack_w_kernel<<<ceil_div(total, 256), 256, 0, stream>>>(W, b, *Wp, *bp,
                                                            Cin, Cout, Cinp, Coutp);
  };
  _Float16* saWp[4][3]; float* saBp[4][3];
  _Float16* fpWp[4][3]; float* fpBp[4][3];
  _Float16* fc1Wp; float* fc1Bp;
  _Float16* fc2Wp; float* fc2Bp;
  int saKp[4][3];
  for (int i = 0; i < 4; ++i)
    for (int l = 0; l < 3; ++l) {
      saKp[i][l] = (l == 0) ? saCinP[i] : saCin[i][l];
      pack(saW[i][l], saB[i][l], saCin[i][l], saCout[i][l], saKp[i][l], saCout[i][l],
           &saWp[i][l], &saBp[i][l]);
    }
  for (int i = 0; i < 4; ++i)
    for (int l = 0; l < fpNL[i]; ++l)
      pack(fpW[i][l], fpB[i][l], fpSpec[i][l], fpSpec[i][l + 1], fpSpec[i][l],
           fpSpec[i][l + 1], &fpWp[i][l], &fpBp[i][l]);
  pack(fc1w, nullptr, 128, 128, 128, 128, &fc1Wp, &fc1Bp);
  pack(fc2w, fc2b, 128, NCLS, 128, 64, &fc2Wp, &fc2Bp);

  // ---- Set Abstraction stages ----
  const float* xyz_prev = pc;
  int n_prev = NPTS;
  for (int i = 0; i < 4; ++i) {
    const int P = NP[i];
    fps_kernel<<<dim3(BATCH), dim3(256), 0, stream>>>(xyz_prev, fpsidx, n_prev, P);
    gather_xyz_kernel<<<ceil_div(BATCH * P, 256), 256, 0, stream>>>(
        xyz_prev, fpsidx, xyzl[i], n_prev, P);
    ball_query_kernel<<<ceil_div(BATCH * P, 256), 256, 0, stream>>>(
        xyzl[i], xyz_prev, ballidx, n_prev, P, RAD[i] * RAD[i]);
    const int Cf = (i == 0) ? 0 : saCout[i - 1][2];
    const _Float16* fprev = (i == 0) ? nullptr : featl[i - 1];
    const int Msa = BATCH * P * NSAMP;
    group_kernel<<<ceil_div(Msa, 256), 256, 0, stream>>>(
        xyz_prev, fprev, xyzl[i], ballidx, bufA, n_prev, P, Cf, saCinP[i]);
    const _Float16* Xc = bufA;
    _Float16* Yc = bufB;
    for (int l = 0; l < 3; ++l) {
      Yc = (l & 1) ? bufA : bufB;
      run_mlp(saWp[i][l], saBp[i][l], Xc, Yc, saKp[i][l], saCout[i][l], Msa, 1, stream);
      Xc = Yc;
    }
    maxpool_kernel<<<ceil_div(BATCH * P * saCout[i][2], 256), 256, 0, stream>>>(
        Xc, featl[i], saCout[i][2], BATCH * P);
    xyz_prev = xyzl[i];
    n_prev = P;
  }

  // ---- Feature Propagation stages (deepest first) ----
  const int fpCk[4] = {128, 256, 256, 512};  // interpolated (known) channels
  const int fpCu[4] = {0, 64, 128, 256};     // skip (unknown) channels
  for (int lvl = 3; lvl >= 0; --lvl) {
    const int nu = (lvl == 0) ? NPTS : NP[lvl - 1];
    const int nk = NP[lvl];
    const float* uxyz = (lvl == 0) ? pc : xyzl[lvl - 1];
    const float* kxyz = xyzl[lvl];
    const _Float16* kfeat = (lvl == 3) ? featl[3] : fpf[lvl + 1];
    const _Float16* ufeat = (lvl == 0) ? nullptr : featl[lvl - 1];
    const int Mfp = BATCH * nu;
    three_nn_kernel<<<ceil_div(Mfp, 256), 256, 0, stream>>>(uxyz, kxyz, nnidx, nnw, nu, nk);
    interp_concat_kernel<<<ceil_div(Mfp, 128), 128, 0, stream>>>(
        kfeat, ufeat, nnidx, nnw, bufA, nu, nk, fpCk[lvl], fpCu[lvl]);
    const _Float16* Xc = bufA;
    for (int l = 0; l < fpNL[lvl]; ++l) {
      _Float16* Yc = (l == fpNL[lvl] - 1) ? fpf[lvl] : ((l & 1) ? bufA : bufB);
      run_mlp(fpWp[lvl][l], fpBp[lvl][l], Xc, Yc, fpSpec[lvl][l], fpSpec[lvl][l + 1],
              Mfp, 1, stream);
      Xc = Yc;
    }
  }

  // ---- Head: fc1 (no bias, relu) -> fc2 (+bias, no relu) -> unpack to f32 ----
  run_mlp(fc1Wp, fc1Bp, fpf[0], bufA, 128, 128, BATCH * NPTS, 1, stream);
  run_mlp(fc2Wp, fc2Bp, bufA, bufB, 128, 64, BATCH * NPTS, 0, stream);
  head_transpose_kernel<<<ceil_div(BATCH * NPTS * NCLS, 256), 256, 0, stream>>>(
      bufB, (float*)d_out, BATCH * NPTS, 64);
}
